// DyHeadBlock_81269371175385
// MI455X (gfx1250) — matrix operand order repaired and checked by reading, not compiled, not run on hardware
//
#include <hip/hip_runtime.h>
#include <hip/hip_bf16.h>

typedef __attribute__((ext_vector_type(16))) __bf16 v16bf;
typedef __attribute__((ext_vector_type(8)))  __bf16 v8bf;
typedef __attribute__((ext_vector_type(8)))  float  v8f;

__device__ __forceinline__ unsigned short f2bf(float f) {
  unsigned int u = __float_as_uint(f);
  u += 0x7FFFu + ((u >> 16) & 1u);          // round-to-nearest-even
  return (unsigned short)(u >> 16);
}
__device__ __forceinline__ int imin(int a, int b){ return a < b ? a : b; }
__device__ __forceinline__ int imax(int a, int b){ return a > b ? a : b; }

// ---------------------------------------------------------------------------
// Repack conv weight (Cout, 256, 3, 3) f32 -> bf16 Wb[(kk*Mpad + o)*256 + c]
// ---------------------------------------------------------------------------
__global__ __launch_bounds__(256) void k_repack(const float* __restrict__ w,
                                                unsigned short* __restrict__ Wb,
                                                int Cout, int Mpad) {
  int idx = blockIdx.x * 256 + threadIdx.x;
  int total = 9 * Mpad * 256;
  if (idx >= total) return;
  int kk  = idx / (Mpad * 256);
  int rem = idx - kk * Mpad * 256;
  int o   = rem >> 8;
  int c   = rem & 255;
  float v = (o < Cout) ? w[((o << 8) + c) * 9 + kk] : 0.f;
  Wb[idx] = f2bf(v);
}

// ---------------------------------------------------------------------------
// Fused deformable-conv GEMM (also plain conv3x3 when om==nullptr).
//   out[b, o, p] = sum_{c,kk} Wb[o, c, kk] * bilinear(x[b,c], tap kk @ p) * mask
// Workgroup: 256 threads (8 waves). Tile: 128(M) x 32(N), K = 9*256 in steps
// of 32 via v_wmma_f32_16x16x32_bf16. grid = (B*Ho*Wo/32, Mpad/128).
// ---------------------------------------------------------------------------
__global__ __launch_bounds__(256) void k_mdcn(
    const float* __restrict__ x, int Hi, int Wi,
    const float* __restrict__ om, int presig, int stride,
    const unsigned short* __restrict__ Wb, int Mpad, int Mout,
    const float* __restrict__ bias,
    float* __restrict__ out, int Ho, int Wo)
{
  __shared__ float meta_w[32][9][4];
  __shared__ int   meta_o[32][9][4];
  __shared__ int   nbase[32];
  __shared__ __attribute__((aligned(32))) unsigned short S[32][272]; // 544B rows (17*32)

  const int tid = threadIdx.x;
  const int HW  = Ho * Wo;
  const int HWi = Hi * Wi;
  const int n0  = blockIdx.x * 32;
  const int Ntot = 2 * HW;    // B = 2

  if (tid < 32) {
    int n = n0 + tid;
    int b = (n < Ntot) ? (n / HW) : 0;
    nbase[tid] = b * 256 * HWi;
  }
  // per-(pixel, tap) bilinear metadata: weights fold mask & validity
  for (int item = tid; item < 32 * 9; item += 256) {
    int nl = item / 9, kk = item - nl * 9;
    int n = n0 + nl;
    float wq[4]; int oq[4];
    if (n < Ntot) {
      int b = n / HW, p = n - b * HW;
      int ho = p / Wo, wo = p - ho * Wo;
      float dy = 0.f, dx = 0.f, m = 1.f;
      if (om) {
        const float* omb = om + (long)b * 27 * HW + p;
        dy = omb[(2 * kk) * HW];
        dx = omb[(2 * kk + 1) * HW];
        float mv = omb[(18 + kk) * HW];
        m = presig ? mv : (1.f / (1.f + __expf(-mv)));
      }
      float sy = (float)(ho * stride - 1 + (kk / 3)) + dy;
      float sx = (float)(wo * stride - 1 + (kk % 3)) + dx;
      float y0f = floorf(sy), x0f = floorf(sx);
      int y0 = (int)y0f, x0 = (int)x0f;
      float wy1 = sy - y0f, wx1 = sx - x0f;
      float wy0 = 1.f - wy1, wx0 = 1.f - wx1;
#pragma unroll
      for (int j = 0; j < 4; ++j) {
        int yi = y0 + (j >> 1), xi = x0 + (j & 1);
        bool v = (yi >= 0) && (yi < Hi) && (xi >= 0) && (xi < Wi);
        int yc = imin(imax(yi, 0), Hi - 1), xc = imin(imax(xi, 0), Wi - 1);
        float w = ((j >> 1) ? wy1 : wy0) * ((j & 1) ? wx1 : wx0);
        wq[j] = v ? (w * m) : 0.f;
        oq[j] = yc * Wi + xc;
      }
    } else {
#pragma unroll
      for (int j = 0; j < 4; ++j) { wq[j] = 0.f; oq[j] = 0; }
    }
#pragma unroll
    for (int j = 0; j < 4; ++j) { meta_w[nl][kk][j] = wq[j]; meta_o[nl][kk][j] = oq[j]; }
  }
  __syncthreads();

  const int lane = tid & 31, wave = tid >> 5;
  const int lm = lane & 15, lh = lane >> 4;
  const int m0 = blockIdx.y * 128 + wave * 16;
  v8f acc0 = {0.f,0.f,0.f,0.f,0.f,0.f,0.f,0.f};
  v8f acc1 = {0.f,0.f,0.f,0.f,0.f,0.f,0.f,0.f};
  const int kA = lh ? 16 : 0;    // contiguous A k order chosen here (matches B): lanes<16 K0..15, lanes>=16 K16..31
  const int kB = lh ? 16 : 0;    // contiguous B k order chosen here (not the ISA table order): lanes<16 K0..15, lanes>=16 K16..31
  const __bf16* WbB = (const __bf16*)Wb;

  const int snl = tid & 31, cg = tid >> 5;
  const float* __restrict__ xb = x + nbase[snl];

  for (int kk = 0; kk < 9; ++kk) {
    // build bf16 S[pixel][channel] tile for this tap: 4-corner gather + FMA
    const float w0 = meta_w[snl][kk][0], w1 = meta_w[snl][kk][1];
    const float w2 = meta_w[snl][kk][2], w3 = meta_w[snl][kk][3];
    const int   o0 = meta_o[snl][kk][0], o1 = meta_o[snl][kk][1];
    const int   o2 = meta_o[snl][kk][2], o3 = meta_o[snl][kk][3];
#pragma unroll 4
    for (int e = 0; e < 32; ++e) {
      int c = cg + (e << 3);
      const float* xp = xb + c * HWi;
      float v = w0 * xp[o0] + w1 * xp[o1] + w2 * xp[o2] + w3 * xp[o3];
      S[snl][c] = f2bf(v);
    }
    __syncthreads();
    if (m0 < Mpad) {
      const __bf16* wrow = WbB + (size_t)(kk * Mpad + (m0 + lm)) * 256 + kA;
#pragma unroll
      for (int cs = 0; cs < 8; ++cs) {
        const int c0 = cs * 32;
        v16bf a = *(const v16bf*)(wrow + c0);
        v16bf b0 = *(const v16bf*)&S[lm     ][c0 + kB];
        v16bf b1 = *(const v16bf*)&S[lm + 16][c0 + kB];
        acc0 = __builtin_amdgcn_wmma_f32_16x16x32_bf16(false, a, false, b0,
                                                       (short)0, acc0, false, false);
        acc1 = __builtin_amdgcn_wmma_f32_16x16x32_bf16(false, a, false, b1,
                                                       (short)0, acc1, false, false);
      }
    }
    __syncthreads();
  }

  if (m0 < Mpad) {
#pragma unroll
    for (int sub = 0; sub < 2; ++sub) {
      v8f acc = sub ? acc1 : acc0;
      int nl = sub * 16 + lm;
      int n = n0 + nl;
      if (n < Ntot) {
        int b = n / HW, p = n - b * HW;
#pragma unroll
        for (int r = 0; r < 8; ++r) {
          int o = m0 + r + lh * 8;   // C/D layout: VGPR r -> M = r (+8 for upper half-wave)
          if (o < Mout) {
            float v = acc[r] + (bias ? bias[o] : 0.f);
            out[((long)b * Mout + o) * HW + p] = v;
          }
        }
      }
    }
  }
}

// ---------------------------------------------------------------------------
// Align-corners bilinear resize of the 27-ch offset/mask map.
// Channels 18..26 get sigmoid BEFORE interpolation (matches reference order).
// ---------------------------------------------------------------------------
__global__ __launch_bounds__(256) void k_resize_om(const float* __restrict__ om,
                                                   float* __restrict__ rom,
                                                   int Hs, int Ws, int Hd, int Wd) {
  int idx = blockIdx.x * 256 + threadIdx.x;
  int total = 2 * 27 * Hd * Wd;
  if (idx >= total) return;
  int wd = idx % Wd;
  int hd = (idx / Wd) % Hd;
  int ch = (idx / (Wd * Hd)) % 27;
  int b  = idx / (27 * Hd * Wd);
  float sy = hd * (float)(Hs - 1) / (float)(Hd - 1);
  float sx = wd * (float)(Ws - 1) / (float)(Wd - 1);
  float y0f = floorf(sy), x0f = floorf(sx);
  int y0 = (int)y0f, x0 = (int)x0f;
  float wy1 = sy - y0f, wx1 = sx - x0f, wy0 = 1.f - wy1, wx0 = 1.f - wx1;
  int y1 = imin(y0 + 1, Hs - 1), x1 = imin(x0 + 1, Ws - 1);
  y0 = imin(y0, Hs - 1); x0 = imin(x0, Ws - 1);
  const float* src = om + ((long)b * 27 + ch) * Hs * Ws;
  float v00 = src[y0 * Ws + x0], v01 = src[y0 * Ws + x1];
  float v10 = src[y1 * Ws + x0], v11 = src[y1 * Ws + x1];
  if (ch >= 18) {  // mask channels: sigmoid first
    v00 = 1.f / (1.f + __expf(-v00)); v01 = 1.f / (1.f + __expf(-v01));
    v10 = 1.f / (1.f + __expf(-v10)); v11 = 1.f / (1.f + __expf(-v11));
  }
  rom[idx] = wy0 * wx0 * v00 + wy0 * wx1 * v01 + wy1 * wx0 * v10 + wy1 * wx1 * v11;
}

// ---------------------------------------------------------------------------
// Per-(b,c) sum / sum-of-squares over H*W (feeds GroupNorm + scale attention)
// ---------------------------------------------------------------------------
__global__ __launch_bounds__(256) void k_stats(const float* __restrict__ Y,
                                               float* __restrict__ sums,
                                               float* __restrict__ sumsq, int HW) {
  int bc = blockIdx.x;
  const float* p = Y + (long)bc * HW;
  float s = 0.f, q = 0.f;
  for (int i = threadIdx.x; i < HW; i += 256) { float v = p[i]; s += v; q += v * v; }
  __shared__ float t1[256], t2[256];
  t1[threadIdx.x] = s; t2[threadIdx.x] = q;
  __syncthreads();
  for (int o2 = 128; o2 > 0; o2 >>= 1) {
    if (threadIdx.x < o2) { t1[threadIdx.x] += t1[threadIdx.x + o2];
                            t2[threadIdx.x] += t2[threadIdx.x + o2]; }
    __syncthreads();
  }
  if (threadIdx.x == 0) { sums[bc] = t1[0]; sumsq[bc] = t2[0]; }
}

// ---------------------------------------------------------------------------
// GroupNorm params (mu, rsig per (b, group)) + scale-attention scalar per b.
// Attention only needs channel means: a = hsig(relu(sum_c gmean_n[c]*w_attn[c]+b))
// ---------------------------------------------------------------------------
__global__ __launch_bounds__(256) void k_scalar(const float* __restrict__ sums,
                                                const float* __restrict__ sumsq,
                                                const float* __restrict__ gamma,
                                                const float* __restrict__ beta,
                                                const float* __restrict__ w_attn,
                                                const float* __restrict__ b_attn,
                                                float* __restrict__ mu_o,
                                                float* __restrict__ rs_o,
                                                float* __restrict__ a_o, int HW) {
  int b = blockIdx.x, c = threadIdx.x;
  __shared__ float t1[256], t2[256];
  __shared__ float mug[16], rsg[16];
  float s = sums[b * 256 + c], q = sumsq[b * 256 + c];
  t1[c] = s; t2[c] = q;
  __syncthreads();
  if (c < 16) {
    float gs = 0.f, gq = 0.f;
    for (int i = 0; i < 16; ++i) { gs += t1[c * 16 + i]; gq += t2[c * 16 + i]; }
    float inv = 1.f / (16.f * (float)HW);
    float mu = gs * inv;
    float var = gq * inv - mu * mu;
    float rs = rsqrtf(var + 1e-5f);
    mug[c] = mu; rsg[c] = rs;
    mu_o[b * 16 + c] = mu; rs_o[b * 16 + c] = rs;
  }
  __syncthreads();
  float cm = (s / (float)HW - mug[c >> 4]) * rsg[c >> 4] * gamma[c] + beta[c];
  t1[c] = cm * w_attn[c];
  __syncthreads();
  for (int o2 = 128; o2 > 0; o2 >>= 1) {
    if (c < o2) t1[c] += t1[c + o2];
    __syncthreads();
  }
  if (c == 0) {
    float a = fmaxf(t1[0] + b_attn[0], 0.f);
    a_o[b] = fminf(fmaxf((a + 3.f) / 6.f, 0.f), 1.f);
  }
}

// ---------------------------------------------------------------------------
// Fused epilogue: GN-normalize mid/low, bilinear-upsample+GN-normalize high,
// attention-weighted sum, /n, ReLU.
// ---------------------------------------------------------------------------
__global__ __launch_bounds__(256) void k_out(
    const float* __restrict__ Ym, const float* muM, const float* rsM,
    const float* gM, const float* bM, const float* aM,
    const float* __restrict__ Yl, const float* muL, const float* rsL,
    const float* gL, const float* bL, const float* aL,
    const float* __restrict__ Yh, const float* muH, const float* rsH,
    const float* gH, const float* bH, const float* aH,
    float* __restrict__ out, int H, int W, int Hh, int Wh, float invn)
{
  int idx = blockIdx.x * 256 + threadIdx.x;
  int HW = H * W;
  int total = 2 * 256 * HW;
  if (idx >= total) return;
  int p = idx % HW;
  int c = (idx / HW) & 255;
  int b = idx / (256 * HW);
  int g = c >> 4;
  float acc;
  {
    float v = Ym[idx];
    acc = ((v - muM[b * 16 + g]) * rsM[b * 16 + g] * gM[c] + bM[c]) * aM[b];
  }
  if (Yl) {
    float v = Yl[idx];
    acc += ((v - muL[b * 16 + g]) * rsL[b * 16 + g] * gL[c] + bL[c]) * aL[b];
  }
  if (Yh) {
    int h = p / W, w = p - h * W;
    float sy = h * (float)(Hh - 1) / (float)(H - 1);
    float sx = w * (float)(Wh - 1) / (float)(W - 1);
    float y0f = floorf(sy), x0f = floorf(sx);
    int y0 = (int)y0f, x0 = (int)x0f;
    float wy1 = sy - y0f, wx1 = sx - x0f, wy0 = 1.f - wy1, wx0 = 1.f - wx1;
    int y1 = imin(y0 + 1, Hh - 1), x1 = imin(x0 + 1, Wh - 1);
    y0 = imin(y0, Hh - 1); x0 = imin(x0, Wh - 1);
    const float* src = Yh + ((long)b * 256 + c) * Hh * Wh;
    float v = wy0 * wx0 * src[y0 * Wh + x0] + wy0 * wx1 * src[y0 * Wh + x1]
            + wy1 * wx0 * src[y1 * Wh + x0] + wy1 * wx1 * src[y1 * Wh + x1];
    acc += ((v - muH[b * 16 + g]) * rsH[b * 16 + g] * gH[c] + bH[c]) * aH[b];
  }
  out[idx] = fmaxf(acc * invn, 0.f);
}

// ---------------------------------------------------------------------------
extern "C" void kernel_launch(void* const* d_in, const int* in_sizes, int n_in,
                              void* d_out, int out_size, void* d_ws, size_t ws_size,
                              hipStream_t stream) {
  const float* x0      = (const float*)d_in[0];
  const float* x1      = (const float*)d_in[1];
  const float* x2      = (const float*)d_in[2];
  const float* w_off   = (const float*)d_in[3];
  const float* b_off   = (const float*)d_in[4];
  const float* w_mid   = (const float*)d_in[5];
  const float* g_mid   = (const float*)d_in[6];
  const float* be_mid  = (const float*)d_in[7];
  const float* w_low   = (const float*)d_in[8];
  const float* g_low   = (const float*)d_in[9];
  const float* be_low  = (const float*)d_in[10];
  const float* w_high  = (const float*)d_in[11];
  const float* g_high  = (const float*)d_in[12];
  const float* be_high = (const float*)d_in[13];
  const float* w_attn  = (const float*)d_in[14];
  const float* b_attn  = (const float*)d_in[15];

  const int HW0 = 80 * 80, HW1 = 40 * 40, HW2 = 20 * 20;

  size_t off = 0;
  auto take = [&](size_t bytes) -> void* {
    off = (off + 255) & ~(size_t)255;
    void* p = (char*)d_ws + off;
    off += bytes;
    return p;
  };

  unsigned short* WbOff  = (unsigned short*)take((size_t)9 * 32  * 256 * 2);
  unsigned short* WbMid  = (unsigned short*)take((size_t)9 * 256 * 256 * 2);
  unsigned short* WbLow  = (unsigned short*)take((size_t)9 * 256 * 256 * 2);
  unsigned short* WbHigh = (unsigned short*)take((size_t)9 * 256 * 256 * 2);
  float* om0  = (float*)take((size_t)2 * 27 * HW0 * 4);
  float* om1  = (float*)take((size_t)2 * 27 * HW1 * 4);
  float* om2  = (float*)take((size_t)2 * 27 * HW2 * 4);
  float* rom1 = (float*)take((size_t)2 * 27 * HW1 * 4);
  float* rom2 = (float*)take((size_t)2 * 27 * HW2 * 4);
  float* Ymid0  = (float*)take((size_t)2 * 256 * HW0 * 4);
  float* Yhigh0 = (float*)take((size_t)2 * 256 * HW1 * 4);
  float* Ymid1  = (float*)take((size_t)2 * 256 * HW1 * 4);
  float* Ylow1  = (float*)take((size_t)2 * 256 * HW1 * 4);
  float* Yhigh1 = (float*)take((size_t)2 * 256 * HW2 * 4);
  float* Ymid2  = (float*)take((size_t)2 * 256 * HW2 * 4);
  float* Ylow2  = (float*)take((size_t)2 * 256 * HW2 * 4);
  // 7 instances of stats: sums[512], sumsq[512], mu[32], rs[32], a[2]
  float *sums[7], *sq[7], *mu[7], *rs[7], *aa[7];
  for (int i = 0; i < 7; ++i) {
    sums[i] = (float*)take(512 * 4);
    sq[i]   = (float*)take(512 * 4);
    mu[i]   = (float*)take(32 * 4);
    rs[i]   = (float*)take(32 * 4);
    aa[i]   = (float*)take(2 * 4);
  }

  // 1) repack weights to bf16 [kk][o][c]
  k_repack<<<(9 * 32 * 256 + 255) / 256, 256, 0, stream>>>(w_off, WbOff, 27, 32);
  k_repack<<<(9 * 256 * 256 + 255) / 256, 256, 0, stream>>>(w_mid, WbMid, 256, 256);
  k_repack<<<(9 * 256 * 256 + 255) / 256, 256, 0, stream>>>(w_low, WbLow, 256, 256);
  k_repack<<<(9 * 256 * 256 + 255) / 256, 256, 0, stream>>>(w_high, WbHigh, 256, 256);

  // 2) offset conv3x3 per level (mdcn with zero offsets / unit mask)
  k_mdcn<<<dim3(2 * HW0 / 32, 1), 256, 0, stream>>>(x0, 80, 80, nullptr, 0, 1,
                                                    WbOff, 32, 27, b_off, om0, 80, 80);
  k_mdcn<<<dim3(2 * HW1 / 32, 1), 256, 0, stream>>>(x1, 40, 40, nullptr, 0, 1,
                                                    WbOff, 32, 27, b_off, om1, 40, 40);
  k_mdcn<<<dim3(2 * HW2 / 32, 1), 256, 0, stream>>>(x2, 20, 20, nullptr, 0, 1,
                                                    WbOff, 32, 27, b_off, om2, 20, 20);

  // 3) resized offset/mask maps for the "high" branches
  k_resize_om<<<(2 * 27 * HW1 + 255) / 256, 256, 0, stream>>>(om0, rom1, 80, 80, 40, 40);
  k_resize_om<<<(2 * 27 * HW2 + 255) / 256, 256, 0, stream>>>(om1, rom2, 40, 40, 20, 20);

  // 4) the 7 MDCN GEMMs (WMMA bf16)
  k_mdcn<<<dim3(2 * HW0 / 32, 2), 256, 0, stream>>>(x0, 80, 80, om0, 0, 1,
                                                    WbMid, 256, 256, nullptr, Ymid0, 80, 80);
  k_mdcn<<<dim3(2 * HW1 / 32, 2), 256, 0, stream>>>(x1, 40, 40, rom1, 1, 1,
                                                    WbHigh, 256, 256, nullptr, Yhigh0, 40, 40);
  k_mdcn<<<dim3(2 * HW1 / 32, 2), 256, 0, stream>>>(x1, 40, 40, om1, 0, 1,
                                                    WbMid, 256, 256, nullptr, Ymid1, 40, 40);
  k_mdcn<<<dim3(2 * HW1 / 32, 2), 256, 0, stream>>>(x0, 80, 80, om1, 0, 2,
                                                    WbLow, 256, 256, nullptr, Ylow1, 40, 40);
  k_mdcn<<<dim3(2 * HW2 / 32, 2), 256, 0, stream>>>(x2, 20, 20, rom2, 1, 1,
                                                    WbHigh, 256, 256, nullptr, Yhigh1, 20, 20);
  k_mdcn<<<dim3(2 * HW2 / 32, 2), 256, 0, stream>>>(x2, 20, 20, om2, 0, 1,
                                                    WbMid, 256, 256, nullptr, Ymid2, 20, 20);
  k_mdcn<<<dim3(2 * HW2 / 32, 2), 256, 0, stream>>>(x1, 40, 40, om2, 0, 2,
                                                    WbLow, 256, 256, nullptr, Ylow2, 20, 20);

  // 5) GroupNorm stats + attention scalars per instance
  const float* Ys[7]  = {Ymid0, Yhigh0, Ymid1, Ylow1, Yhigh1, Ymid2, Ylow2};
  const int    HWs[7] = {HW0, HW1, HW1, HW1, HW2, HW2, HW2};
  const float* Gs[7]  = {g_mid, g_high, g_mid, g_low, g_high, g_mid, g_low};
  const float* Bs[7]  = {be_mid, be_high, be_mid, be_low, be_high, be_mid, be_low};
  for (int i = 0; i < 7; ++i) {
    k_stats<<<512, 256, 0, stream>>>(Ys[i], sums[i], sq[i], HWs[i]);
    k_scalar<<<2, 256, 0, stream>>>(sums[i], sq[i], Gs[i], Bs[i], w_attn, b_attn,
                                    mu[i], rs[i], aa[i], HWs[i]);
  }

  // 6) fused normalize + upsample + attention-weighted combine + ReLU
  float* out0 = (float*)d_out;
  float* out1 = out0 + (size_t)2 * 256 * HW0;
  float* out2 = out1 + (size_t)2 * 256 * HW1;
  // level 0: mid0 + up(high0), n=2
  k_out<<<(2 * 256 * HW0 + 255) / 256, 256, 0, stream>>>(
      Ymid0, mu[0], rs[0], g_mid, be_mid, aa[0],
      nullptr, nullptr, nullptr, nullptr, nullptr, nullptr,
      Yhigh0, mu[1], rs[1], g_high, be_high, aa[1],
      out0, 80, 80, 40, 40, 0.5f);
  // level 1: mid1 + low1 + up(high1), n=3
  k_out<<<(2 * 256 * HW1 + 255) / 256, 256, 0, stream>>>(
      Ymid1, mu[2], rs[2], g_mid, be_mid, aa[2],
      Ylow1, mu[3], rs[3], g_low, be_low, aa[3],
      Yhigh1, mu[4], rs[4], g_high, be_high, aa[4],
      out1, 40, 40, 20, 20, 1.f / 3.f);
  // level 2: mid2 + low2, n=2
  k_out<<<(2 * 256 * HW2 + 255) / 256, 256, 0, stream>>>(
      Ymid2, mu[5], rs[5], g_mid, be_mid, aa[5],
      Ylow2, mu[6], rs[6], g_low, be_low, aa[6],
      nullptr, nullptr, nullptr, nullptr, nullptr, nullptr,
      out2, 20, 20, 1, 1, 0.5f);
  (void)in_sizes; (void)n_in; (void)out_size; (void)ws_size;
}
